// CombinedGraphReadout_24910810316918
// MI455X (gfx1250) — matrix operand order repaired and checked by reading, not compiled, not run on hardware
//
#include <hip/hip_runtime.h>
#include <math.h>

#define N_NODES  131072
#define N_GRAPHS 8192
#define NODE_DIM 256
#define HID      512
#define NHEADS   16
#define OUT_DIM  512
#define RAW_DIM  (3*OUT_DIM)   // 1536

typedef __attribute__((ext_vector_type(16))) __bf16 v16bf;
typedef __attribute__((ext_vector_type(8)))  __bf16 v8bf;
typedef __attribute__((ext_vector_type(8)))  float  v8f;

__device__ __forceinline__ v8f wmma_bf16(v16bf a, v16bf b, v8f c) {
  // D = A(16x32 bf16) * B(32x16 bf16) + C(16x16 f32)
  return __builtin_amdgcn_wmma_f32_16x16x32_bf16(false, a, false, b, (short)0, c, false, false);
}

__device__ __forceinline__ v16bf combine8(v8bf lo, v8bf hi) {
  return __builtin_shufflevector(lo, hi, 0,1,2,3,4,5,6,7,8,9,10,11,12,13,14,15);
}

// A-fragment (16x32) of a row-major f32 matrix, converted to bf16.
// Layout (ISA 7.12.2): lanes 0-15 -> M=lane, K = k0+[0..7] and k0+[16..23];
// lanes 16-31 -> M=lane-16, K = k0+[8..15] and k0+[24..31].
__device__ __forceinline__ v16bf load_a_f32(const float* __restrict__ X, int ld,
                                            int row0, int k0, int lane) {
  const int half = lane >> 4, m = lane & 15;
  const float* p = X + (size_t)(row0 + m) * ld + k0 + half * 8;
  v8bf lo, hi;
#pragma unroll
  for (int i = 0; i < 8; i++) lo[i] = (__bf16)p[i];
#pragma unroll
  for (int i = 0; i < 8; i++) hi[i] = (__bf16)p[16 + i];
  return combine8(lo, hi);
}

// Same, but map -inf (empty-graph segment max) to 0 like the reference.
__device__ __forceinline__ v16bf load_a_f32_clamp(const float* __restrict__ X, int ld,
                                                  int row0, int k0, int lane) {
  const int half = lane >> 4, m = lane & 15;
  const float* p = X + (size_t)(row0 + m) * ld + k0 + half * 8;
  v8bf lo, hi;
#pragma unroll
  for (int i = 0; i < 8; i++) { float f = p[i];      if (f < -1e30f) f = 0.f; lo[i] = (__bf16)f; }
#pragma unroll
  for (int i = 0; i < 8; i++) { float f = p[16 + i]; if (f < -1e30f) f = 0.f; hi[i] = (__bf16)f; }
  return combine8(lo, hi);
}

// A-fragment from a row-major bf16 LDS tile.
__device__ __forceinline__ v16bf load_a_lds(const __bf16* H, int ld, int k0, int lane) {
  const int half = lane >> 4, m = lane & 15;
  const __bf16* p = H + m * ld + k0 + half * 8;
  v8bf lo = *(const v8bf*)p;
  v8bf hi = *(const v8bf*)(p + 16);
  return combine8(lo, hi);
}

// B-fragment (32x16) from transposed bf16 weights WT[N][K] (row per output col).
// Lanes 0-15 -> col=lane, K=k0+[0..7],k0+[16..23] ; lanes 16-31 -> col=lane-16, K=k0+[8..15],k0+[24..31].
__device__ __forceinline__ v16bf load_b(const __bf16* __restrict__ WT, int ldk,
                                        int n0, int k0, int lane) {
  const int half = lane >> 4, n = lane & 15;
  const __bf16* p = WT + (size_t)(n0 + n) * ldk + k0 + half * 8;
  return combine8(*(const v8bf*)p, *(const v8bf*)(p + 16));
}

// C frag (16x16 f32): lane -> col (lane&15), VGPR v -> row v + 8*(lane>=16).
__device__ __forceinline__ void store_c_bias_relu_lds(v8f acc, const float* bias, int n0,
                                                      __bf16* H, int ld, int lane) {
  const int col = n0 + (lane & 15);
  const float b = bias[col];
  const int mbase = (lane >> 4) << 3;
#pragma unroll
  for (int v = 0; v < 8; v++) {
    float f = acc[v] + b;
    f = f > 0.f ? f : 0.f;
    H[(mbase + v) * ld + col] = (__bf16)f;
  }
}

__device__ __forceinline__ void store_c_relu_lds(v8f acc, int n0, __bf16* H, int ld, int lane) {
  const int col = n0 + (lane & 15);
  const int mbase = (lane >> 4) << 3;
#pragma unroll
  for (int v = 0; v < 8; v++) {
    float f = acc[v];
    f = f > 0.f ? f : 0.f;
    H[(mbase + v) * ld + col] = (__bf16)f;
  }
}

// ---------------------------------------------------------------- utility kernels

__global__ void k_fill(float* __restrict__ p, size_t n, float v) {
  size_t i = (size_t)blockIdx.x * blockDim.x + threadIdx.x;
  if (i < n) p[i] = v;
}

// dst[n*K + k] = bf16(src[k*N + n])   (transpose + downconvert)
__global__ void k_transpose(const float* __restrict__ src, __bf16* __restrict__ dst,
                            int K, int N) {
  size_t i = (size_t)blockIdx.x * blockDim.x + threadIdx.x;
  if (i >= (size_t)K * N) return;
  int k = (int)(i / N), n = (int)(i % N);
  dst[(size_t)n * K + k] = (__bf16)src[i];
}

__global__ void k_segmax_x(const float* __restrict__ X, const int* __restrict__ gid,
                           float* __restrict__ Mx) {
  size_t i = (size_t)blockIdx.x * blockDim.x + threadIdx.x;
  if (i >= (size_t)N_NODES * NODE_DIM) return;
  int node = (int)(i >> 8), c = (int)(i & 255);
  __hip_atomic_fetch_max(&Mx[(size_t)gid[node] * NODE_DIM + c], X[i],
                         __ATOMIC_RELAXED, __HIP_MEMORY_SCOPE_AGENT);
}

__global__ void k_expnorm(float* __restrict__ exm, const float* __restrict__ segmax,
                          float* __restrict__ segsum, const int* __restrict__ gid) {
  size_t i = (size_t)blockIdx.x * blockDim.x + threadIdx.x;
  if (i >= (size_t)N_NODES * NHEADS) return;
  int node = (int)(i >> 4), h = (int)(i & 15);
  size_t gi = (size_t)gid[node] * NHEADS + h;
  float e = __expf(exm[i] - segmax[gi]);
  exm[i] = e;
  __hip_atomic_fetch_add(&segsum[gi], e, __ATOMIC_RELAXED, __HIP_MEMORY_SCOPE_AGENT);
}

// ---------------------------------------------------------------- K1: score MLPs
// One wave per 32-node tile: two 16-row M-tiles share each B fragment
// (halves weight traffic, 2 WMMAs per B load).

__global__ __launch_bounds__(32) void k_scores(
    const float* __restrict__ X, const int* __restrict__ gid,
    const __bf16* __restrict__ mW1T, const float* __restrict__ mb1,
    const __bf16* __restrict__ mW2T, const float* __restrict__ mb2,
    const __bf16* __restrict__ sW1T, const float* __restrict__ sb1,
    const __bf16* __restrict__ sW2T, const float* __restrict__ sb2,
    float* __restrict__ exm, float* __restrict__ wsig, float* __restrict__ segmax) {
  __shared__ __bf16 Hs[32 * HID];                 // 32 KB hidden tile (32 rows)
  const int lane = threadIdx.x & 31;
  const int row0 = blockIdx.x * 32;

  v16bf ax[2][8];
#pragma unroll
  for (int t = 0; t < 2; t++)
#pragma unroll
    for (int k = 0; k < 8; k++)
      ax[t][k] = load_a_f32(X, NODE_DIM, row0 + t * 16, k * 32, lane);

  const __bf16* W1T[2] = {mW1T, sW1T};
  const float*  B1 [2] = {mb1,  sb1};
  const __bf16* W2T[2] = {mW2T, sW2T};
  const float*  B2 [2] = {mb2,  sb2};

  for (int br = 0; br < 2; br++) {
    for (int nt = 0; nt < HID / 16; nt++) {
      v8f acc0 = {}, acc1 = {};
#pragma unroll
      for (int k = 0; k < 8; k++) {
        v16bf b = load_b(W1T[br], NODE_DIM, nt * 16, k * 32, lane);
        acc0 = wmma_bf16(ax[0][k], b, acc0);
        acc1 = wmma_bf16(ax[1][k], b, acc1);
      }
      store_c_bias_relu_lds(acc0, B1[br], nt * 16, Hs, HID, lane);
      store_c_bias_relu_lds(acc1, B1[br], nt * 16, Hs + 16 * HID, HID, lane);
    }
    v8f sc[2] = {{}, {}};
#pragma unroll
    for (int k = 0; k < 16; k++) {
      v16bf b = load_b(W2T[br], HID, 0, k * 32, lane);
      sc[0] = wmma_bf16(load_a_lds(Hs, HID, k * 32, lane), b, sc[0]);
      sc[1] = wmma_bf16(load_a_lds(Hs + 16 * HID, HID, k * 32, lane), b, sc[1]);
    }
    const int h = lane & 15;
    const float b2 = B2[br][h];
#pragma unroll
    for (int t = 0; t < 2; t++) {
#pragma unroll
      for (int v = 0; v < 8; v++) {
        int node = row0 + t * 16 + v + ((lane >> 4) << 3);
        float s = sc[t][v] + b2;
        if (br == 0) {
          exm[(size_t)node * NHEADS + h] = s;
          __hip_atomic_fetch_max(&segmax[(size_t)gid[node] * NHEADS + h], s,
                                 __ATOMIC_RELAXED, __HIP_MEMORY_SCOPE_AGENT);
        } else {
          wsig[(size_t)node * NHEADS + h] = 1.f / (1.f + __expf(-s));
        }
      }
    }
  }
}

// ---------------------------------------------------------------- K3: value MLPs + weighted scatter

__global__ __launch_bounds__(32) void k_values(
    const float* __restrict__ X, const int* __restrict__ gid,
    const __bf16* __restrict__ mW1T, const float* __restrict__ mb1,
    const __bf16* __restrict__ mW2T, const float* __restrict__ mb2,
    const __bf16* __restrict__ sW1T, const float* __restrict__ sb1,
    const __bf16* __restrict__ sW2T, const float* __restrict__ sb2,
    const float* __restrict__ exm, const float* __restrict__ segsum,
    const float* __restrict__ wsig,
    float* __restrict__ Pm, float* __restrict__ Ps) {
  __shared__ __bf16 Hs[32 * HID];                 // 32 KB
  __shared__ float  Wt[2][32 * NHEADS];           // 4 KB per-node per-head weights
  const int lane = threadIdx.x & 31;
  const int row0 = blockIdx.x * 32;

  for (int i = lane; i < 32 * NHEADS; i += 32) {
    int m = i >> 4, h = i & 15, node = row0 + m, g = gid[node];
    Wt[0][i] = exm[(size_t)node * NHEADS + h] / segsum[(size_t)g * NHEADS + h];
    Wt[1][i] = wsig[(size_t)node * NHEADS + h];
  }

  v16bf ax[2][8];
#pragma unroll
  for (int t = 0; t < 2; t++)
#pragma unroll
    for (int k = 0; k < 8; k++)
      ax[t][k] = load_a_f32(X, NODE_DIM, row0 + t * 16, k * 32, lane);

  const int g_lo = gid[row0], g_hi = gid[row0 + 31];
  const bool uni = (g_lo == g_hi);

  const __bf16* W1T[2] = {mW1T, sW1T};
  const float*  B1 [2] = {mb1,  sb1};
  const __bf16* W2T[2] = {mW2T, sW2T};
  const float*  B2 [2] = {mb2,  sb2};
  float* P[2] = {Pm, Ps};

  for (int br = 0; br < 2; br++) {
    for (int nt = 0; nt < HID / 16; nt++) {
      v8f acc0 = {}, acc1 = {};
#pragma unroll
      for (int k = 0; k < 8; k++) {
        v16bf b = load_b(W1T[br], NODE_DIM, nt * 16, k * 32, lane);
        acc0 = wmma_bf16(ax[0][k], b, acc0);
        acc1 = wmma_bf16(ax[1][k], b, acc1);
      }
      store_c_bias_relu_lds(acc0, B1[br], nt * 16, Hs, HID, lane);
      store_c_bias_relu_lds(acc1, B1[br], nt * 16, Hs + 16 * HID, HID, lane);
    }
    const float* Wgt = Wt[br];
    for (int nt = 0; nt < HID / 16; nt++) {
      v8f vc[2] = {{}, {}};
#pragma unroll
      for (int k = 0; k < 16; k++) {
        v16bf b = load_b(W2T[br], HID, nt * 16, k * 32, lane);
        vc[0] = wmma_bf16(load_a_lds(Hs, HID, k * 32, lane), b, vc[0]);
        vc[1] = wmma_bf16(load_a_lds(Hs + 16 * HID, HID, k * 32, lane), b, vc[1]);
      }
      const int n = lane & 15, ncol = nt * 16 + n, head = nt >> 1;
      const float b2 = B2[br][ncol];
      if (uni) {
        float part = 0.f;
#pragma unroll
        for (int t = 0; t < 2; t++)
#pragma unroll
          for (int v = 0; v < 8; v++) {
            int m = t * 16 + v + ((lane >> 4) << 3);
            part += Wgt[m * NHEADS + head] * (vc[t][v] + b2);
          }
        __hip_atomic_fetch_add(&P[br][(size_t)g_lo * HID + ncol], part,
                               __ATOMIC_RELAXED, __HIP_MEMORY_SCOPE_AGENT);
      } else {
#pragma unroll
        for (int t = 0; t < 2; t++)
#pragma unroll
          for (int v = 0; v < 8; v++) {
            int m = t * 16 + v + ((lane >> 4) << 3);
            float c = Wgt[m * NHEADS + head] * (vc[t][v] + b2);
            __hip_atomic_fetch_add(&P[br][(size_t)gid[row0 + m] * HID + ncol], c,
                                   __ATOMIC_RELAXED, __HIP_MEMORY_SCOPE_AGENT);
          }
      }
    }
  }
}

// ---------------------------------------------------------------- K4: per-graph combine

__global__ __launch_bounds__(32) void k_combine(
    const float* __restrict__ Pm, const float* __restrict__ Ps, const float* __restrict__ Mx,
    const __bf16* __restrict__ mWcT, const __bf16* __restrict__ sWcT,
    const __bf16* __restrict__ WmaxT, const __bf16* __restrict__ WfinT,
    float* __restrict__ out) {
  __shared__ __bf16 R[16 * RAW_DIM];              // 48 KB relu(concat) tile
  const int lane = threadIdx.x & 31;
  const int g0 = blockIdx.x * 16;

  // mean_repr = Pm @ mWc    -> R cols [0,512)
  for (int nt = 0; nt < OUT_DIM / 16; nt++) {
    v8f acc = {};
#pragma unroll
    for (int k = 0; k < 16; k++)
      acc = wmma_bf16(load_a_f32(Pm, HID, g0, k * 32, lane),
                      load_b(mWcT, HID, nt * 16, k * 32, lane), acc);
    store_c_relu_lds(acc, nt * 16, R, RAW_DIM, lane);
  }
  // sum_repr = Ps @ sWc     -> R cols [512,1024)
  for (int nt = 0; nt < OUT_DIM / 16; nt++) {
    v8f acc = {};
#pragma unroll
    for (int k = 0; k < 16; k++)
      acc = wmma_bf16(load_a_f32(Ps, HID, g0, k * 32, lane),
                      load_b(sWcT, HID, nt * 16, k * 32, lane), acc);
    store_c_relu_lds(acc, OUT_DIM + nt * 16, R, RAW_DIM, lane);
  }
  // max_repr = clamp(Mx) @ Wmax -> R cols [1024,1536)
  for (int nt = 0; nt < OUT_DIM / 16; nt++) {
    v8f acc = {};
#pragma unroll
    for (int k = 0; k < 8; k++)
      acc = wmma_bf16(load_a_f32_clamp(Mx, NODE_DIM, g0, k * 32, lane),
                      load_b(WmaxT, NODE_DIM, nt * 16, k * 32, lane), acc);
    store_c_relu_lds(acc, 2 * OUT_DIM + nt * 16, R, RAW_DIM, lane);
  }
  // out = relu(raw) @ Wfinal
  for (int nt = 0; nt < OUT_DIM / 16; nt++) {
    v8f acc = {};
#pragma unroll
    for (int k = 0; k < RAW_DIM / 32; k++)
      acc = wmma_bf16(load_a_lds(R, RAW_DIM, k * 32, lane),
                      load_b(WfinT, RAW_DIM, nt * 16, k * 32, lane), acc);
    const int ncol = nt * 16 + (lane & 15);
#pragma unroll
    for (int v = 0; v < 8; v++) {
      int m = v + ((lane >> 4) << 3);
      out[(size_t)(g0 + m) * OUT_DIM + ncol] = acc[v];
    }
  }
}

// ---------------------------------------------------------------- host launcher

extern "C" void kernel_launch(void* const* d_in, const int* in_sizes, int n_in,
                              void* d_out, int out_size, void* d_ws, size_t ws_size,
                              hipStream_t stream) {
  const float* X    = (const float*)d_in[0];
  const int*   gid  = (const int*)d_in[1];
  const float* mWs1 = (const float*)d_in[3];  const float* mbs1 = (const float*)d_in[4];
  const float* mWs2 = (const float*)d_in[5];  const float* mbs2 = (const float*)d_in[6];
  const float* mWt1 = (const float*)d_in[7];  const float* mbt1 = (const float*)d_in[8];
  const float* mWt2 = (const float*)d_in[9];  const float* mbt2 = (const float*)d_in[10];
  const float* mWc  = (const float*)d_in[11];
  const float* sWs1 = (const float*)d_in[12]; const float* sbs1 = (const float*)d_in[13];
  const float* sWs2 = (const float*)d_in[14]; const float* sbs2 = (const float*)d_in[15];
  const float* sWt1 = (const float*)d_in[16]; const float* sbt1 = (const float*)d_in[17];
  const float* sWt2 = (const float*)d_in[18]; const float* sbt2 = (const float*)d_in[19];
  const float* sWc  = (const float*)d_in[20];
  const float* Wmax = (const float*)d_in[21];
  const float* Wfin = (const float*)d_in[22];
  float* out = (float*)d_out;

  char* ws = (char*)d_ws;
  size_t off = 0;
  auto bump = [&](size_t bytes) -> char* {
    char* p = ws + off; off = (off + bytes + 255) & ~(size_t)255; return p;
  };
  __bf16* mWs1T = (__bf16*)bump((size_t)HID * NODE_DIM * 2);
  __bf16* mWs2T = (__bf16*)bump((size_t)NHEADS * HID * 2);
  __bf16* mWt1T = (__bf16*)bump((size_t)HID * NODE_DIM * 2);
  __bf16* mWt2T = (__bf16*)bump((size_t)HID * HID * 2);
  __bf16* sWs1T = (__bf16*)bump((size_t)HID * NODE_DIM * 2);
  __bf16* sWs2T = (__bf16*)bump((size_t)NHEADS * HID * 2);
  __bf16* sWt1T = (__bf16*)bump((size_t)HID * NODE_DIM * 2);
  __bf16* sWt2T = (__bf16*)bump((size_t)HID * HID * 2);
  __bf16* mWcT  = (__bf16*)bump((size_t)OUT_DIM * HID * 2);
  __bf16* sWcT  = (__bf16*)bump((size_t)OUT_DIM * HID * 2);
  __bf16* WmaxT = (__bf16*)bump((size_t)OUT_DIM * NODE_DIM * 2);
  __bf16* WfinT = (__bf16*)bump((size_t)OUT_DIM * RAW_DIM * 2);
  float* exm    = (float*)bump((size_t)N_NODES * NHEADS * 4);
  float* wsig   = (float*)bump((size_t)N_NODES * NHEADS * 4);
  float* segmax = (float*)bump((size_t)N_GRAPHS * NHEADS * 4);
  float* segsum = (float*)bump((size_t)N_GRAPHS * NHEADS * 4);
  float* Mx     = (float*)bump((size_t)N_GRAPHS * NODE_DIM * 4);
  float* Pm     = (float*)bump((size_t)N_GRAPHS * HID * 4);
  float* Ps     = (float*)bump((size_t)N_GRAPHS * HID * 4);

  const int TB = 256;
  auto cdiv = [](size_t a, size_t b) -> unsigned { return (unsigned)((a + b - 1) / b); };

  // init accumulators
  k_fill<<<cdiv((size_t)N_GRAPHS * NHEADS, TB), TB, 0, stream>>>(segmax, (size_t)N_GRAPHS * NHEADS, -INFINITY);
  k_fill<<<cdiv((size_t)N_GRAPHS * NHEADS, TB), TB, 0, stream>>>(segsum, (size_t)N_GRAPHS * NHEADS, 0.f);
  k_fill<<<cdiv((size_t)N_GRAPHS * NODE_DIM, TB), TB, 0, stream>>>(Mx, (size_t)N_GRAPHS * NODE_DIM, -INFINITY);
  k_fill<<<cdiv((size_t)N_GRAPHS * HID, TB), TB, 0, stream>>>(Pm, (size_t)N_GRAPHS * HID, 0.f);
  k_fill<<<cdiv((size_t)N_GRAPHS * HID, TB), TB, 0, stream>>>(Ps, (size_t)N_GRAPHS * HID, 0.f);

  // weight transpose + bf16 downconvert
  struct { const float* s; __bf16* d; int K, N; } tp[12] = {
    {mWs1, mWs1T, NODE_DIM, HID}, {mWs2, mWs2T, HID, NHEADS},
    {mWt1, mWt1T, NODE_DIM, HID}, {mWt2, mWt2T, HID, HID},
    {sWs1, sWs1T, NODE_DIM, HID}, {sWs2, sWs2T, HID, NHEADS},
    {sWt1, sWt1T, NODE_DIM, HID}, {sWt2, sWt2T, HID, HID},
    {mWc,  mWcT,  HID, OUT_DIM},  {sWc,  sWcT,  HID, OUT_DIM},
    {Wmax, WmaxT, NODE_DIM, OUT_DIM}, {Wfin, WfinT, RAW_DIM, OUT_DIM},
  };
  for (int i = 0; i < 12; i++)
    k_transpose<<<cdiv((size_t)tp[i].K * tp[i].N, TB), TB, 0, stream>>>(tp[i].s, tp[i].d, tp[i].K, tp[i].N);

  // per-graph max of node embeddings
  k_segmax_x<<<cdiv((size_t)N_NODES * NODE_DIM, TB), TB, 0, stream>>>(X, gid, Mx);

  // scores (mean branch raw scores + softmax max; sum branch sigmoid)
  k_scores<<<N_NODES / 32, 32, 0, stream>>>(X, gid, mWs1T, mbs1, mWs2T, mbs2,
                                            sWs1T, sbs1, sWs2T, sbs2, exm, wsig, segmax);
  // softmax exp + segment sum
  k_expnorm<<<cdiv((size_t)N_NODES * NHEADS, TB), TB, 0, stream>>>(exm, segmax, segsum, gid);

  // value MLPs + weighted segment sums
  k_values<<<N_NODES / 32, 32, 0, stream>>>(X, gid, mWt1T, mbt1, mWt2T, mbt2,
                                            sWt1T, sbt1, sWt2T, sbt2,
                                            exm, segsum, wsig, Pm, Ps);
  // per-graph combine + final projection
  k_combine<<<N_GRAPHS / 16, 32, 0, stream>>>(Pm, Ps, Mx, mWcT, sWcT, WmaxT, WfinT, out);
}
